// EdgeConvNet_37314675868438
// MI455X (gfx1250) — matrix operand order repaired and checked by reading, not compiled, not run on hardware
//
#include <hip/hip_runtime.h>
#include <hip/hip_bf16.h>

typedef __attribute__((ext_vector_type(16))) _Float16 v16h;
typedef __attribute__((ext_vector_type(8)))  float    v8f;
typedef __attribute__((ext_vector_type(2)))  float    f32x2;

// Each wave processes 2 edges per iteration:
//   A (16x64 f16, as two 16x32 tiles): row m = (edge_in_pair*8 + o),
//       A[m,k] = relu(conv_w[o,0]*f1[k] + conv_w[o,1]*f2[k] + conv_b[o])
//   B (64x16 f16, two 32x16 tiles)   : B[k,n] = w1[n,k]   (built once)
//   D = A*B (16x16 f32) -> fc1 pre-bias; fc2/fc3 via multi-value butterfly.
__global__ __launch_bounds__(256, 4)
void edgeconv_wmma_kernel(const float* __restrict__ features,
                          const long long* __restrict__ edge_index,
                          const float* __restrict__ conv_w,
                          const float* __restrict__ conv_b,
                          const float* __restrict__ w1,
                          const float* __restrict__ b1,
                          const float* __restrict__ w2,
                          const float* __restrict__ b2,
                          const float* __restrict__ w3,
                          const float* __restrict__ b3,
                          float* __restrict__ out,
                          int nEdges)
{
    const int lane   = threadIdx.x & 31;
    const int wavesPerBlock = blockDim.x >> 5;
    const int waveId = blockIdx.x * wavesPerBlock + (threadIdx.x >> 5);
    const int nWaves = gridDim.x * wavesPerBlock;

    const int half = lane >> 4;   // which 16-lane half of the wave
    const int m    = lane & 15;   // A-row index / D-column index (=g)
    const int oRow = m & 7;       // conv output channel of this A row
    const int eSel = m >> 3;      // which edge of the pair owns this A row

    // ---- per-lane uniform constants ----
    const float aw    = conv_w[2 * oRow + 0];
    const float bw    = conv_w[2 * oRow + 1];
    const float cb    = conv_b[oRow];
    const float b1g   = b1[m];     // fc1 bias for column g=m (added post-WMMA)
    const float w2g   = w2[m];     // fc2 weight for column g=m
    const float bias2 = b2[0];
    const float bias3 = b3[0];
    // After the multi-value reduction, this lane holds the fc2 sum of row:
    const int   rowId = (lane >> 1) & 7;
    const float w3row = w3[rowId];

    // ---- build B = w1^T once (two 32x16 f16 tiles) ----
    // B-fragment: lane holds column n=m; VGPR v holds K = 8*half + 2v + j (v<4), 16 + 8*half + 2(v-4) + j (v>=4).
    v16h btile0, btile1;
    {
        const float* w1row = w1 + (size_t)m * 64;
#pragma unroll
        for (int v = 0; v < 8; ++v) {
            const int kb = (v < 4) ? (half * 8 + 2 * v)
                                   : (16 + half * 8 + 2 * (v - 4));
            f32x2 q0 = *(const f32x2*)(w1row + kb);        // k in [0,32)
            f32x2 q1 = *(const f32x2*)(w1row + 32 + kb);   // k in [32,64)
            btile0[2 * v + 0] = (_Float16)q0[0];
            btile0[2 * v + 1] = (_Float16)q0[1];
            btile1[2 * v + 0] = (_Float16)q1[0];
            btile1[2 * v + 1] = (_Float16)q1[1];
        }
    }

    const int nGroups = (nEdges + 1) >> 1;   // 2 edges per group

    for (int grp = waveId; grp < nGroups; grp += nWaves) {
        const int e0  = grp * 2;
        const int e1  = (e0 + 1 < nEdges) ? (e0 + 1) : (nEdges - 1); // clamp tail
        const int eA  = eSel ? e1 : e0;      // edge feeding this lane's A rows

        // prefetch next group's edge indices (global_prefetch_b8)
        const int nxt = grp + nWaves;
        if (nxt < nGroups) {
            __builtin_prefetch(&edge_index[(size_t)nxt * 2], 0, 3);
            __builtin_prefetch(&edge_index[(size_t)nEdges + (size_t)nxt * 2], 0, 3);
        }

        const long long s = edge_index[eA];
        const long long d = edge_index[(size_t)nEdges + eA];
        const float* f1 = features + (size_t)s * 64;
        const float* f2 = features + (size_t)d * 64;

        // ---- fused gather + conv + relu directly into A fragments ----
        // A-fragment: lane holds row m; VGPR v holds K pair at kb (see ISA).
        v16h a0, a1;
#pragma unroll
        for (int v = 0; v < 8; ++v) {
            const int kb = (v < 4) ? (half * 8 + 2 * v)
                                   : (16 + half * 8 + 2 * (v - 4));
            f32x2 x0 = *(const f32x2*)(f1 + kb);
            f32x2 y0 = *(const f32x2*)(f2 + kb);
            f32x2 x1 = *(const f32x2*)(f1 + 32 + kb);
            f32x2 y1 = *(const f32x2*)(f2 + 32 + kb);
            float t;
            t = fmaf(aw, x0[0], fmaf(bw, y0[0], cb)); a0[2*v+0] = (_Float16)fmaxf(t, 0.0f);
            t = fmaf(aw, x0[1], fmaf(bw, y0[1], cb)); a0[2*v+1] = (_Float16)fmaxf(t, 0.0f);
            t = fmaf(aw, x1[0], fmaf(bw, y1[0], cb)); a1[2*v+0] = (_Float16)fmaxf(t, 0.0f);
            t = fmaf(aw, x1[1], fmaf(bw, y1[1], cb)); a1[2*v+1] = (_Float16)fmaxf(t, 0.0f);
        }

        // ---- fc1 as WMMA: D = A(16x64) * w1^T(64x16), f32 accumulate ----
        v8f c = {};
        c = __builtin_amdgcn_wmma_f32_16x16x32_f16(false, a0, false, btile0,
                                                   (short)0, c, false, false);
        c = __builtin_amdgcn_wmma_f32_16x16x32_f16(false, a1, false, btile1,
                                                   (short)0, c, false, false);

        // ---- fc2 partials: lane holds column g=m of rows r=0..7 (its half's edge)
        float p[8];
#pragma unroll
        for (int r = 0; r < 8; ++r)
            p[r] = fmaxf(c[r] + b1g, 0.0f) * w2g;

        // ---- multi-value butterfly: 8 rows x 16 lanes -> row sums, 8 shuffles
        // xor 8: keep 4 rows, trade the other 4 with the partner
        float q4[4];
#pragma unroll
        for (int j = 0; j < 4; ++j) {
            const float keep = (lane & 8) ? p[j + 4] : p[j];
            const float send = (lane & 8) ? p[j]     : p[j + 4];
            q4[j] = keep + __shfl_xor(send, 8, 32);
        }
        // xor 4: keep 2
        float q2[2];
#pragma unroll
        for (int j = 0; j < 2; ++j) {
            const float keep = (lane & 4) ? q4[j + 2] : q4[j];
            const float send = (lane & 4) ? q4[j]     : q4[j + 2];
            q2[j] = keep + __shfl_xor(send, 4, 32);
        }
        // xor 2: keep 1
        {
            const float keep = (lane & 2) ? q2[1] : q2[0];
            const float send = (lane & 2) ? q2[0] : q2[1];
            q2[0] = keep + __shfl_xor(send, 2, 32);
        }
        // xor 1: finish the 16-lane sum; lane now holds fc2 sum of row rowId
        const float fc2sum = q2[0] + __shfl_xor(q2[0], 1, 32);

        // ---- fc2 bias+relu, fc3 weight, then reduce over rows (lane bits 1..3)
        float part = fmaxf(fc2sum + bias2, 0.0f) * w3row;
        part += __shfl_xor(part, 2, 32);
        part += __shfl_xor(part, 4, 32);
        part += __shfl_xor(part, 8, 32);

        const float sig = 1.0f / (1.0f + __expf(-(part + bias3)));

        // lanes 0-15 own edge e0; lanes 16-31 own edge e0+1
        const int eOut = e0 + half;
        if (m == 0 && eOut < nEdges) out[eOut] = sig;
    }
}

extern "C" void kernel_launch(void* const* d_in, const int* in_sizes, int n_in,
                              void* d_out, int out_size, void* d_ws, size_t ws_size,
                              hipStream_t stream) {
    // setup_inputs order:
    // 0:x 1:features 2:edge_index 3:conv_w 4:conv_b 5:w1 6:b1 7:w2 8:b2 9:w3 10:b3
    const float*     features   = (const float*)d_in[1];
    const long long* edge_index = (const long long*)d_in[2];
    const float*     conv_w     = (const float*)d_in[3];
    const float*     conv_b     = (const float*)d_in[4];
    const float*     w1         = (const float*)d_in[5];
    const float*     b1         = (const float*)d_in[6];
    const float*     w2         = (const float*)d_in[7];
    const float*     b2         = (const float*)d_in[8];
    const float*     w3         = (const float*)d_in[9];
    const float*     b3         = (const float*)d_in[10];
    float* out = (float*)d_out;

    const int nEdges = in_sizes[2] / 2;

    const int threads = 256;            // 8 wave32 per block
    const int blocks  = 1024;           // 8192 waves, ~49 edge-pairs each
    edgeconv_wmma_kernel<<<blocks, threads, 0, stream>>>(
        features, edge_index, conv_w, conv_b, w1, b1, w2, b2, w3, b3,
        out, nEdges);
}
